// FlashSigmoidCrossAttention_5085241278762
// MI455X (gfx1250) — matrix operand order repaired and checked by reading, not compiled, not run on hardware
//
#include <hip/hip_runtime.h>

// ---------------- problem constants (match reference) ----------------
constexpr int B_   = 2;
constexpr int SQ_  = 2048;
constexpr int SKV_ = 2048;
constexpr int D_   = 1024;
constexpr int H_   = 16;
constexpr int HD_  = 64;
constexpr float SIGMOID_BIAS = -8.0f;
constexpr float LN_EPS = 1e-5f;

// ---------------- WMMA types ----------------
typedef __attribute__((ext_vector_type(16))) __bf16 v16bf;
typedef __attribute__((ext_vector_type(8)))  float  v8f;

union FragBF {             // one bf16 WMMA operand (8 VGPRs, 16 bf16/lane)
    v16bf v;
    uint4 q[2];
    unsigned u[8];
};
union AccF {               // 16x16 f32 accumulator (8 VGPRs)
    v8f   v;
    float f[8];
};

static __device__ __forceinline__ v8f wmma_bf16(v16bf a, v16bf b, v8f c) {
    return __builtin_amdgcn_wmma_f32_16x16x32_bf16(false, a, false, b, (short)0, c, false, false);
}

// round-to-nearest-even f32 -> bf16
static __device__ __forceinline__ unsigned short f2bf(float x) {
    unsigned u = __builtin_bit_cast(unsigned, x);
    u += 0x7FFFu + ((u >> 16) & 1u);
    return (unsigned short)(u >> 16);
}
static __device__ __forceinline__ unsigned f2bf_pack(float a, float b) {
    unsigned ua = __builtin_bit_cast(unsigned, a);
    unsigned ub = __builtin_bit_cast(unsigned, b);
    ua += 0x7FFFu + ((ua >> 16) & 1u);
    ub += 0x7FFFu + ((ub >> 16) & 1u);
    return (ua >> 16) | (ub & 0xFFFF0000u);
}

// fast sigmoid: v_exp_f32 + v_rcp_f32 (no IEEE division chain)
static __device__ __forceinline__ float fast_sigmoid(float x) {
    return __builtin_amdgcn_rcpf(1.0f + __expf(-x));
}

// ---------------- kernel A: elementwise f32 -> bf16 (8 elems / thread) ----------------
__global__ __launch_bounds__(256) void k_cvt_bf16(const float* __restrict__ in,
                                                  unsigned short* __restrict__ out) {
    unsigned i = blockIdx.x * 256 + threadIdx.x;
    const float4* p = (const float4*)in + (size_t)i * 2;
    float4 f0 = p[0], f1 = p[1];
    uint4 o;
    o.x = f2bf_pack(f0.x, f0.y);
    o.y = f2bf_pack(f0.z, f0.w);
    o.z = f2bf_pack(f1.x, f1.y);
    o.w = f2bf_pack(f1.z, f1.w);
    ((uint4*)out)[i] = o;
}

// ---------------- kernel B: f32 [R x C] -> bf16 transposed [C x R] ----------------
__global__ __launch_bounds__(256) void k_transpose_bf16(const float* __restrict__ in,
                                                        unsigned short* __restrict__ out,
                                                        int R, int C) {
    __shared__ float tile[32][33];
    int c = blockIdx.x * 32 + threadIdx.x;
    #pragma unroll
    for (int i = 0; i < 4; ++i) {
        int r = blockIdx.y * 32 + threadIdx.y + i * 8;
        if (r < R && c < C) tile[threadIdx.y + i * 8][threadIdx.x] = in[(size_t)r * C + c];
    }
    __syncthreads();
    int x = blockIdx.y * 32 + threadIdx.x;              // original row (contiguous out dim)
    #pragma unroll
    for (int i = 0; i < 4; ++i) {
        int y = blockIdx.x * 32 + threadIdx.y + i * 8;  // original col
        if (x < R && y < C) out[(size_t)y * R + x] = f2bf(tile[threadIdx.x][threadIdx.y + i * 8]);
    }
}

// ---------------- kernel C: C = A(bf16, MxK) @ Bt(bf16, NxK)^T ----------------
// Compile-time strides; per-wave 16x16 C tile, 8 waves = 32x64 block tile.
// Columns < NS  -> bf16(scale*val) into Cb (ld = NS)
// Columns >= NS -> f32 val into Cf (ld = N-NS)
// NS is a multiple of 64 so the split is uniform per block tile.
template<int K, int N, int NS>
__global__ __launch_bounds__(256) void k_gemm_bf16(const unsigned short* __restrict__ A,
                                                   const unsigned short* __restrict__ Bt,
                                                   unsigned short* __restrict__ Cb,
                                                   float* __restrict__ Cf,
                                                   float scale) {
    const int w    = threadIdx.x >> 5;
    const int l    = threadIdx.x & 31;
    const int half = l >> 4;
    const int n    = l & 15;
    const int m0 = blockIdx.x * 32 + (w & 1) * 16;
    const int n0 = blockIdx.y * 64 + (w >> 1) * 16;

    const unsigned short* ap = A  + (unsigned)(m0 + n) * K;   // both lane-halves: same row
    const unsigned short* bp = Bt + (unsigned)(n0 + n) * K;

    v8f acc = {};
    for (int k0 = 0; k0 < K; k0 += 64) {
        if (k0 + 64 < K) {
            __builtin_prefetch(ap + k0 + 64, 0, 1);
            __builtin_prefetch(bp + k0 + 64, 0, 1);
        }
        #pragma unroll
        for (int s = 0; s < 2; ++s) {
            const int kk = k0 + s * 32;
            FragBF a, b;
            a.q[0] = *(const uint4*)(ap + kk + half * 8);
            a.q[1] = *(const uint4*)(ap + kk + 16 + half * 8);
            b.q[0] = *(const uint4*)(bp + kk + half * 8);
            b.q[1] = *(const uint4*)(bp + kk + 16 + half * 8);
            acc = wmma_bf16(a.v, b.v, acc);
        }
    }

    AccF r; r.v = acc;
    const int nn = n0 + n;
    if constexpr (NS == N) {
        #pragma unroll
        for (int e = 0; e < 8; ++e)
            Cb[(unsigned)(m0 + e + 8 * half) * NS + nn] = f2bf(r.f[e] * scale);
    } else if constexpr (NS == 0) {
        #pragma unroll
        for (int e = 0; e < 8; ++e)
            Cf[(unsigned)(m0 + e + 8 * half) * N + nn] = r.f[e];
    } else {
        if (n0 < NS) {        // uniform across the wave
            #pragma unroll
            for (int e = 0; e < 8; ++e)
                Cb[(unsigned)(m0 + e + 8 * half) * NS + nn] = f2bf(r.f[e] * scale);
        } else {
            #pragma unroll
            for (int e = 0; e < 8; ++e)
                Cf[(unsigned)(m0 + e + 8 * half) * (N - NS) + (nn - NS)] = r.f[e];
        }
    }
}

// ---------------- kernel D: LayerNorm over HD=64 per (b,s,h); transposed bf16 out ----------------
// out layout: vT[((b*H + h)*HD + hd) * SKV + s]
__global__ __launch_bounds__(256) void k_ln_v(const float* __restrict__ vraw,
                                              const float* __restrict__ gamma,
                                              const float* __restrict__ beta,
                                              unsigned short* __restrict__ vT) {
    const int w = threadIdx.x >> 5;
    const int l = threadIdx.x & 31;
    const long row = (long)blockIdx.x * 8 + w;          // over B*SKV*H
    const int m = (int)(row >> 4);                      // b*SKV + s
    const int h = (int)(row & 15);
    const float* p = vraw + (size_t)m * D_ + h * HD_;
    float x0 = p[l], x1 = p[l + 32];
    float s = x0 + x1;
    #pragma unroll
    for (int off = 16; off >= 1; off >>= 1) s += __shfl_xor(s, off, 32);
    float mu = s * (1.0f / 64.0f);
    float d0 = x0 - mu, d1 = x1 - mu;
    float vv = d0 * d0 + d1 * d1;
    #pragma unroll
    for (int off = 16; off >= 1; off >>= 1) vv += __shfl_xor(vv, off, 32);
    float rinv = rsqrtf(vv * (1.0f / 64.0f) + LN_EPS);
    int b = m / SKV_, ss = m % SKV_;
    size_t base = ((size_t)(b * H_ + h) * HD_) * SKV_ + ss;
    vT[base + (size_t)l * SKV_]        = f2bf(d0 * rinv * gamma[l]      + beta[l]);
    vT[base + (size_t)(l + 32) * SKV_] = f2bf(d1 * rinv * gamma[l + 32] + beta[l + 32]);
}

// ---------------- kernel E: sigmoid flash attention ----------------
// qs,ks: bf16 [B,S,D] pre-scaled by HD^-0.25 ; vT: bf16 [B,H,HD,SKV]
// out_av: f32 [B,SQ,D] ; av_bf: bf16 copy for projection GEMM
__global__ __launch_bounds__(256) void k_attn(const unsigned short* __restrict__ qs,
                                              const unsigned short* __restrict__ ks,
                                              const unsigned short* __restrict__ vT,
                                              float* __restrict__ out_av,
                                              unsigned short* __restrict__ av_bf) {
    __shared__ __align__(16) unsigned short attn_lds[8][16 * 32];

    const int w    = threadIdx.x >> 5;
    const int l    = threadIdx.x & 31;
    const int half = l >> 4;
    const int n    = l & 15;
    const int h  = blockIdx.y;
    const int b  = blockIdx.z;
    const int q0 = blockIdx.x * 128 + w * 16;

    // ---- q A-fragments: 16 rows x 64 K as two 16x32 chunks ----
    FragBF aq[2];
    {
        const unsigned short* qp = qs + ((size_t)(b * SQ_ + q0 + n)) * D_ + h * HD_;
        #pragma unroll
        for (int c = 0; c < 2; ++c) {
            aq[c].q[0] = *(const uint4*)(qp + c * 32 + half * 8);
            aq[c].q[1] = *(const uint4*)(qp + c * 32 + 16 + half * 8);
        }
    }

    v8f acc[4] = {v8f{}, v8f{}, v8f{}, v8f{}};
    const int nsteps = ((q0 + 15) >> 5) + 1;            // causal bound: kv <= q0+15

    // incrementing base pointers (avoid per-step 64-bit shifts)
    const unsigned short* kp0 = ks + ((size_t)(b * SKV_ + n)) * D_ + h * HD_;            // row j0+n
    const unsigned short* vb0 = vT + ((size_t)(b * H_ + h) * HD_) * SKV_ + half * 8;     // + j0 later

    for (int st = 0; st < nsteps; ++st) {
        const int j0 = st * 32;

        // ---- logits: two 16x16 tiles; sigmoid+mask -> bf16 into this wave's LDS slab ----
        #pragma unroll
        for (int t = 0; t < 2; ++t) {
            const unsigned short* kp = kp0 + (size_t)(t * 16) * D_;
            v8f lg = {};
            #pragma unroll
            for (int c = 0; c < 2; ++c) {
                FragBF bk;
                bk.q[0] = *(const uint4*)(kp + c * 32 + half * 8);
                bk.q[1] = *(const uint4*)(kp + c * 32 + 16 + half * 8);
                lg = wmma_bf16(aq[c].v, bk.v, lg);
            }
            AccF lr; lr.v = lg;
            #pragma unroll
            for (int e = 0; e < 8; ++e) {
                int mrow = e + 8 * half;
                int qidx = q0 + mrow;
                int kidx = j0 + t * 16 + n;
                float sgm = fast_sigmoid(lr.f[e] + SIGMOID_BIAS);
                float a = (kidx <= qidx) ? sgm : 0.0f;
                attn_lds[w][mrow * 32 + t * 16 + n] = f2bf(a);
            }
        }
        __builtin_amdgcn_wave_barrier();

        // ---- re-stripe attn 16x32 into A-fragment layout (LDS in-order per wave) ----
        FragBF aa;
        aa.q[0] = *(const uint4*)&attn_lds[w][n * 32 + half * 8];
        aa.q[1] = *(const uint4*)&attn_lds[w][n * 32 + 16 + half * 8];
        __builtin_amdgcn_wave_barrier();

        // ---- attn(16x32) @ v(32x64): four 16x16 hd tiles ----
        #pragma unroll
        for (int t = 0; t < 4; ++t) {
            FragBF bv;
            const unsigned short* vp = vb0 + (size_t)(t * 16 + n) * SKV_;
            bv.q[0] = *(const uint4*)vp;
            bv.q[1] = *(const uint4*)(vp + 16);
            acc[t] = wmma_bf16(aa.v, bv.v, acc[t]);
        }

        kp0 += (size_t)32 * D_;   // next 32 kv rows
        vb0 += 32;                // next 32 kv columns of vT
    }

    // ---- store attn_times_v (f32) + bf16 copy ----
    #pragma unroll
    for (int t = 0; t < 4; ++t) {
        AccF r; r.v = acc[t];
        #pragma unroll
        for (int e = 0; e < 8; ++e) {
            int qidx = q0 + e + 8 * half;
            int hd   = t * 16 + n;
            unsigned oi = (unsigned)(b * SQ_ + qidx) * D_ + h * HD_ + hd;
            out_av[oi] = r.f[e];
            av_bf[oi]  = f2bf(r.f[e]);
        }
    }
}

// ---------------- host launch ----------------
extern "C" void kernel_launch(void* const* d_in, const int* in_sizes, int n_in,
                              void* d_out, int out_size, void* d_ws, size_t ws_size,
                              hipStream_t stream) {
    const float* q      = (const float*)d_in[0];
    const float* kv     = (const float*)d_in[1];
    const float* Wq     = (const float*)d_in[2];
    const float* Wkv    = (const float*)d_in[3];
    const float* vgamma = (const float*)d_in[4];
    const float* vbeta  = (const float*)d_in[5];
    const float* Wproj  = (const float*)d_in[6];

    const float scale = 0.35355339059327373f;   // HD^(-1/4) = 64^-0.25

    // workspace layout (bytes)  -- total 72 MB
    char* ws = (char*)d_ws;
    size_t off = 0;
    unsigned short* WqT    = (unsigned short*)(ws + off); off += (size_t)D_ * D_ * 2;          // 2 MB
    unsigned short* WkvT   = (unsigned short*)(ws + off); off += (size_t)2 * D_ * D_ * 2;      // 4 MB
    unsigned short* WprojT = (unsigned short*)(ws + off); off += (size_t)D_ * D_ * 2;          // 2 MB
    unsigned short* q_bf   = (unsigned short*)(ws + off); off += (size_t)B_ * SQ_ * D_ * 2;    // 8 MB
    unsigned short* kv_bf  = (unsigned short*)(ws + off); off += (size_t)B_ * SKV_ * D_ * 2;   // 8 MB
    unsigned short* qh_s   = (unsigned short*)(ws + off); off += (size_t)B_ * SQ_ * D_ * 2;    // 8 MB
    unsigned short* k_s    = (unsigned short*)(ws + off); off += (size_t)B_ * SKV_ * D_ * 2;   // 8 MB
    float*          v_raw  = (float*)        (ws + off); off += (size_t)B_ * SKV_ * D_ * 4;    // 16 MB
    unsigned short* vT     = (unsigned short*)(ws + off); off += (size_t)B_ * SKV_ * D_ * 2;   // 8 MB
    unsigned short* av_bf  = (unsigned short*)(ws + off); off += (size_t)B_ * SQ_ * D_ * 2;    // 8 MB

    float* out_av   = (float*)d_out;                          // attn_times_v [B,SQ,D]
    float* out_proj = (float*)d_out + (size_t)B_ * SQ_ * D_;  // attn_proj   [B,SQ,D]

    const int M = B_ * SQ_;   // 4096

    // 0) activations -> bf16 (removes per-tile reconversion in GEMMs)
    k_cvt_bf16<<<(M * D_) / (256 * 8), 256, 0, stream>>>(q, q_bf);
    k_cvt_bf16<<<(M * D_) / (256 * 8), 256, 0, stream>>>(kv, kv_bf);

    // weight transposes -> bf16 (NT-form GEMMs => contiguous B fragments)
    k_transpose_bf16<<<dim3(D_ / 32, D_ / 32), dim3(32, 8), 0, stream>>>(Wq, WqT, D_, D_);
    k_transpose_bf16<<<dim3(2 * D_ / 32, D_ / 32), dim3(32, 8), 0, stream>>>(Wkv, WkvT, D_, 2 * D_);
    k_transpose_bf16<<<dim3(D_ / 32, D_ / 32), dim3(32, 8), 0, stream>>>(Wproj, WprojT, D_, D_);

    // 1) qh_s = bf16((q @ Wq) * scale)
    k_gemm_bf16<D_, D_, D_><<<dim3(M / 32, D_ / 64), 256, 0, stream>>>(
        q_bf, WqT, qh_s, (float*)nullptr, scale);

    // 2) kv @ Wkv : first D cols -> k_s (bf16*scale), last D cols -> v_raw (f32)
    k_gemm_bf16<D_, 2 * D_, D_><<<dim3(M / 32, 2 * D_ / 64), 256, 0, stream>>>(
        kv_bf, WkvT, k_s, v_raw, scale);

    // 3) layernorm v -> transposed bf16
    k_ln_v<<<(B_ * SKV_ * H_) / 8, 256, 0, stream>>>(v_raw, vgamma, vbeta, vT);

    // 4) attention
    k_attn<<<dim3(SQ_ / 128, H_, B_), 256, 0, stream>>>(qh_s, k_s, vT, out_av, av_bf);

    // 5) projection: attn_proj = av @ Wproj (f32 out)
    k_gemm_bf16<D_, D_, 0><<<dim3(M / 32, D_ / 64), 256, 0, stream>>>(
        av_bf, WprojT, (unsigned short*)nullptr, out_proj, 1.0f);
}
